// CustomAttention_33543694581789
// MI455X (gfx1250) — matrix operand order repaired and checked by reading, not compiled, not run on hardware
//
#include <hip/hip_runtime.h>

// ---------------------------------------------------------------------------
// CDNA5 (gfx1250) fused multi-head attention with relative position bias.
// All GEMMs run through v_wmma_f32_16x16x32_f16 (f16 inputs, f32 accumulate).
// Per-wave GEMM tile is 32(m) x 64(n): 8 x v8f accumulators (64 VGPRs) to stay
// well under the VGPR budget (previous 64x64 tile spilled to scratch).
// ---------------------------------------------------------------------------

typedef _Float16 h8   __attribute__((ext_vector_type(8)));
typedef _Float16 v16h __attribute__((ext_vector_type(16)));
typedef float    v8f  __attribute__((ext_vector_type(8)));

#define EMBED   1024
#define HEADS   16
#define HDIM    64
#define MAXREL  64
#define NREL    129      // 2*MAXREL + 1
#define NRELP   144      // padded to multiple of 16
#define BATCH   4
#define SEQ     1024
#define MROWS   (BATCH * SEQ)   // 4096

#define MI 2             // 16-row tiles per wave  (32 rows)
#define NJ 4             // 16-col tiles per wave  (64 cols)

union F16x16 { v16h v; h8 h[2]; };

// ---- WMMA wrapper ----------------------------------------------------------
__device__ __forceinline__ v8f wmma16(v16h a, v16h b, v8f c) {
  // (neg_a, A, neg_b, B, c_mod, C, reuse_a, reuse_b)
  return __builtin_amdgcn_wmma_f32_16x16x32_f16(false, a, false, b, (short)0, c,
                                                false, false);
}

// ---- Fragment loaders (contiguous k = 16h + s, same for A and B) -----------------------
// A 16x32 f16: lane l<16 -> row m0+l, K {0..15};
//              lane l+16 -> row m0+l, K {16..31}
__device__ __forceinline__ v16h load_a16(const _Float16* __restrict base, int ld,
                                         int m0, int k0, int lane) {
  int row = m0 + (lane & 15);
  int ks  = k0 + ((lane & 16) ? 16 : 0);
  const _Float16* p = base + (size_t)row * ld + ks;
  F16x16 f;
  f.h[0] = *(const h8*)(p);
  f.h[1] = *(const h8*)(p + 8);
  return f.v;
}

// Same A layout but fp32 source, converted on the fly (attn weights).
__device__ __forceinline__ v16h load_a16_f32(const float* __restrict base, int ld,
                                             int m0, int k0, int lane) {
  int row = m0 + (lane & 15);
  int ks  = k0 + ((lane & 16) ? 16 : 0);
  const float* p = base + (size_t)row * ld + ks;
  v16h v;
#pragma unroll
  for (int i = 0; i < 8; ++i) {
    v[i]     = (_Float16)p[i];
    v[8 + i] = (_Float16)p[8 + i];
  }
  return v;
}

// B 32x16 f16 supplied as rows of B^T: lane n<16 -> row n0+n, K {0..15};
//                                      lane n+16 -> row n0+n, K {16..31}
__device__ __forceinline__ v16h load_b16(const _Float16* __restrict baseT, int ld,
                                         int n0, int k0, int lane) {
  int row = n0 + (lane & 15);
  int ks  = k0 + ((lane & 16) ? 16 : 0);
  const _Float16* p = baseT + (size_t)row * ld + ks;
  F16x16 f;
  f.h[0] = *(const h8*)(p);
  f.h[1] = *(const h8*)(p + 8);
  return f.v;
}

// Guarded B loader from fp32 source (rel_pos_embed rows, n < nmax).
__device__ __forceinline__ v16h load_b16_f32g(const float* __restrict baseT, int ld,
                                              int n0, int k0, int lane, int nmax) {
  int row = n0 + (lane & 15);
  int ks  = k0 + ((lane & 16) ? 16 : 0);
  v16h v;
  if (row < nmax) {
    const float* p = baseT + (size_t)row * ld + ks;
#pragma unroll
    for (int i = 0; i < 16; ++i) v[i] = (_Float16)p[i];
  } else {
#pragma unroll
    for (int i = 0; i < 16; ++i) v[i] = (_Float16)0.f;
  }
  return v;
}

// ---- K0: fp32 -> f16 convert ----------------------------------------------
__global__ void cvt_f16_kernel(const float* __restrict src,
                               _Float16* __restrict dst, int n) {
  int i = blockIdx.x * blockDim.x + threadIdx.x;
  if (i < n) dst[i] = (_Float16)src[i];
}

// ---- K0b: weight transpose + convert: WT[n][k] = (f16)W[k][n] --------------
__global__ void transpose_w_kernel(const float* __restrict W,
                                   _Float16* __restrict WT) {
  __shared__ float tile[16][17];
  int tx = threadIdx.x & 15, ty = threadIdx.x >> 4;
  int k0 = blockIdx.x * 16, n0 = blockIdx.y * 16;
  tile[ty][tx] = W[(size_t)(k0 + ty) * EMBED + (n0 + tx)];
  __syncthreads();
  WT[(size_t)(n0 + ty) * EMBED + (k0 + tx)] = (_Float16)tile[tx][ty];
}

// ---- K1: projection GEMM  P = X @ W + b,  head-split f16 output ------------
// mode 0: out[b][h][s][d]   (Q, K)      mode 1: out[b][h][d][s]   (V^T)
__global__ __launch_bounds__(128)
void proj_gemm_kernel(const _Float16* __restrict X,
                      const _Float16* __restrict WT,
                      const float* __restrict bias,
                      _Float16* __restrict out, int mode) {
  int lane = threadIdx.x & 31;
  int wave = blockIdx.x * (blockDim.x >> 5) + (threadIdx.x >> 5);
  const int NT = EMBED / (16 * NJ);          // 16 n-strips
  int m0 = (wave / NT) * (16 * MI);
  int n0 = (wave % NT) * (16 * NJ);

  v8f acc[MI][NJ];
#pragma unroll
  for (int i = 0; i < MI; ++i)
#pragma unroll
    for (int j = 0; j < NJ; ++j) acc[i][j] = (v8f)0.f;

  for (int k0 = 0; k0 < EMBED; k0 += 32) {
    v16h a[MI], b[NJ];
#pragma unroll
    for (int i = 0; i < MI; ++i) a[i] = load_a16(X, EMBED, m0 + 16 * i, k0, lane);
#pragma unroll
    for (int j = 0; j < NJ; ++j) b[j] = load_b16(WT, EMBED, n0 + 16 * j, k0, lane);
#pragma unroll
    for (int i = 0; i < MI; ++i)
#pragma unroll
      for (int j = 0; j < NJ; ++j) acc[i][j] = wmma16(a[i], b[j], acc[i][j]);
  }

#pragma unroll
  for (int i = 0; i < MI; ++i)
#pragma unroll
    for (int j = 0; j < NJ; ++j)
#pragma unroll
      for (int r = 0; r < 8; ++r) {
        int m = m0 + 16 * i + ((lane & 16) ? (r + 8) : r);
        int n = n0 + 16 * j + (lane & 15);
        float val = acc[i][j][r] + bias[n];
        int bb = m >> 10, s = m & (SEQ - 1);
        int h  = n >> 6,  d = n & (HDIM - 1);
        size_t idx = mode
            ? ((((size_t)(bb * HEADS + h)) * HDIM + d) * SEQ + s)
            : ((((size_t)(bb * HEADS + h)) * SEQ + s) * HDIM + d);
        out[idx] = (_Float16)val;
      }
}

// ---- K2: scores = Q K^T / sqrt(D), written fp32 into attn region -----------
__global__ __launch_bounds__(128)
void scores_gemm_kernel(const _Float16* __restrict Q,
                        const _Float16* __restrict K,
                        float* __restrict attn) {
  int lane = threadIdx.x & 31;
  int bh   = blockIdx.y;
  int wave = blockIdx.x * (blockDim.x >> 5) + (threadIdx.x >> 5);
  const int NT = SEQ / (16 * NJ);            // 16 n-strips
  int m0 = (wave / NT) * (16 * MI);
  int n0 = (wave % NT) * (16 * NJ);
  const _Float16* q = Q + (size_t)bh * SEQ * HDIM;
  const _Float16* k = K + (size_t)bh * SEQ * HDIM;
  float* o = attn + (size_t)bh * SEQ * SEQ;

  v8f acc[MI][NJ];
#pragma unroll
  for (int i = 0; i < MI; ++i)
#pragma unroll
    for (int j = 0; j < NJ; ++j) acc[i][j] = (v8f)0.f;

#pragma unroll
  for (int k0 = 0; k0 < HDIM; k0 += 32) {
    v16h a[MI], b[NJ];
#pragma unroll
    for (int i = 0; i < MI; ++i) a[i] = load_a16(q, HDIM, m0 + 16 * i, k0, lane);
#pragma unroll
    for (int j = 0; j < NJ; ++j) b[j] = load_b16(k, HDIM, n0 + 16 * j, k0, lane);
#pragma unroll
    for (int i = 0; i < MI; ++i)
#pragma unroll
      for (int j = 0; j < NJ; ++j) acc[i][j] = wmma16(a[i], b[j], acc[i][j]);
  }

#pragma unroll
  for (int i = 0; i < MI; ++i)
#pragma unroll
    for (int j = 0; j < NJ; ++j)
#pragma unroll
      for (int r = 0; r < 8; ++r) {
        int m = m0 + 16 * i + ((lane & 16) ? (r + 8) : r);
        int n = n0 + 16 * j + (lane & 15);
        o[(size_t)m * SEQ + n] = acc[i][j][r] * 0.125f;   // 1/sqrt(64)
      }
}

// ---- K3: T = Q @ emb^T  (129 distinct rel vectors, padded to 144) ----------
__global__ __launch_bounds__(128)
void relT_gemm_kernel(const _Float16* __restrict Q,
                      const float* __restrict emb,
                      float* __restrict T) {
  int lane = threadIdx.x & 31;
  int bh   = blockIdx.y;
  int wave = blockIdx.x * (blockDim.x >> 5) + (threadIdx.x >> 5);
  int m0   = wave * 16;
  const _Float16* q = Q + (size_t)bh * SEQ * HDIM;
  float* t = T + (size_t)bh * SEQ * NRELP;

  v8f acc[9];
#pragma unroll
  for (int j = 0; j < 9; ++j) acc[j] = (v8f)0.f;

#pragma unroll
  for (int k0 = 0; k0 < HDIM; k0 += 32) {
    v16h a = load_a16(q, HDIM, m0, k0, lane);
#pragma unroll
    for (int j = 0; j < 9; ++j) {
      v16h b = load_b16_f32g(emb, HDIM, j * 16, k0, lane, NREL);
      acc[j] = wmma16(a, b, acc[j]);
    }
  }

#pragma unroll
  for (int j = 0; j < 9; ++j)
#pragma unroll
    for (int r = 0; r < 8; ++r) {
      int m = m0 + ((lane & 16) ? (r + 8) : r);
      int n = j * 16 + (lane & 15);
      t[(size_t)m * NRELP + n] = acc[j][r];
    }
}

// ---- K4: add rel bias + mask + row softmax (in place, fp32) ----------------
__global__ __launch_bounds__(256)
void softmax_rel_kernel(float* __restrict attn,
                        const float* __restrict T,
                        const int* __restrict mask) {
  int i  = blockIdx.x;
  int bh = blockIdx.y;
  int b  = bh >> 4;
  float* row        = attn + ((size_t)bh * SEQ + i) * SEQ;
  const float* trow = T    + ((size_t)bh * SEQ + i) * NRELP;
  const int*   mrow = mask + ((size_t)b  * SEQ + i) * SEQ;

  __shared__ float tl[NRELP];
  __shared__ float red[256];
  int tid = threadIdx.x;
  if (tid < NRELP) tl[tid] = trow[tid];
  __syncthreads();

  float vals[4];
  float mx = -INFINITY;
#pragma unroll
  for (int c = 0; c < 4; ++c) {
    int j = tid + c * 256;
    int diff = i - j;
    if (diff > MAXREL) diff = MAXREL;
    if (diff < -MAXREL) diff = -MAXREL;
    float v = row[j] + tl[diff + MAXREL];
    if (mrow[j] == 0) v = -INFINITY;
    vals[c] = v;
    mx = fmaxf(mx, v);
  }
  red[tid] = mx;
  __syncthreads();
  for (int s = 128; s > 0; s >>= 1) {
    if (tid < s) red[tid] = fmaxf(red[tid], red[tid + s]);
    __syncthreads();
  }
  mx = red[0];
  __syncthreads();

  float sum = 0.f;
#pragma unroll
  for (int c = 0; c < 4; ++c) {
    vals[c] = __expf(vals[c] - mx);
    sum += vals[c];
  }
  red[tid] = sum;
  __syncthreads();
  for (int s = 128; s > 0; s >>= 1) {
    if (tid < s) red[tid] += red[tid + s];
    __syncthreads();
  }
  float inv = 1.f / red[0];
#pragma unroll
  for (int c = 0; c < 4; ++c) row[tid + c * 256] = vals[c] * inv;
}

// ---- K5: ctx = attn @ V  (attn fp32 converted in-loader, V^T f16) ----------
__global__ __launch_bounds__(128)
void pv_gemm_kernel(const float* __restrict attn,
                    const _Float16* __restrict VT,
                    _Float16* __restrict ctx) {
  int lane = threadIdx.x & 31;
  int bh   = blockIdx.y;
  int wave = blockIdx.x * (blockDim.x >> 5) + (threadIdx.x >> 5);
  int m0   = wave * (16 * MI);               // n covers all 64 cols (NJ tiles)
  const float* arow   = attn + (size_t)bh * SEQ * SEQ;
  const _Float16* vt  = VT   + (size_t)bh * HDIM * SEQ;

  v8f acc[MI][NJ];
#pragma unroll
  for (int i = 0; i < MI; ++i)
#pragma unroll
    for (int j = 0; j < NJ; ++j) acc[i][j] = (v8f)0.f;

  for (int k0 = 0; k0 < SEQ; k0 += 32) {
    v16h a[MI], b[NJ];
#pragma unroll
    for (int i = 0; i < MI; ++i) a[i] = load_a16_f32(arow, SEQ, m0 + 16 * i, k0, lane);
#pragma unroll
    for (int j = 0; j < NJ; ++j) b[j] = load_b16(vt, SEQ, j * 16, k0, lane);
#pragma unroll
    for (int i = 0; i < MI; ++i)
#pragma unroll
      for (int j = 0; j < NJ; ++j) acc[i][j] = wmma16(a[i], b[j], acc[i][j]);
  }

  int bb = bh >> 4, h = bh & 15;
#pragma unroll
  for (int i = 0; i < MI; ++i)
#pragma unroll
    for (int j = 0; j < NJ; ++j)
#pragma unroll
      for (int r = 0; r < 8; ++r) {
        int m = m0 + 16 * i + ((lane & 16) ? (r + 8) : r);
        int d = 16 * j + (lane & 15);
        ctx[((size_t)bb * SEQ + m) * EMBED + h * HDIM + d] = (_Float16)acc[i][j][r];
      }
}

// ---- K6: out = ctx @ Wo + bo  (fp32 output) --------------------------------
__global__ __launch_bounds__(128)
void out_gemm_kernel(const _Float16* __restrict ctx,
                     const _Float16* __restrict WoT,
                     const float* __restrict bo,
                     float* __restrict out) {
  int lane = threadIdx.x & 31;
  int wave = blockIdx.x * (blockDim.x >> 5) + (threadIdx.x >> 5);
  const int NT = EMBED / (16 * NJ);
  int m0 = (wave / NT) * (16 * MI);
  int n0 = (wave % NT) * (16 * NJ);

  v8f acc[MI][NJ];
#pragma unroll
  for (int i = 0; i < MI; ++i)
#pragma unroll
    for (int j = 0; j < NJ; ++j) acc[i][j] = (v8f)0.f;

  for (int k0 = 0; k0 < EMBED; k0 += 32) {
    v16h a[MI], b[NJ];
#pragma unroll
    for (int i = 0; i < MI; ++i) a[i] = load_a16(ctx, EMBED, m0 + 16 * i, k0, lane);
#pragma unroll
    for (int j = 0; j < NJ; ++j) b[j] = load_b16(WoT, EMBED, n0 + 16 * j, k0, lane);
#pragma unroll
    for (int i = 0; i < MI; ++i)
#pragma unroll
      for (int j = 0; j < NJ; ++j) acc[i][j] = wmma16(a[i], b[j], acc[i][j]);
  }

#pragma unroll
  for (int i = 0; i < MI; ++i)
#pragma unroll
    for (int j = 0; j < NJ; ++j)
#pragma unroll
      for (int r = 0; r < 8; ++r) {
        int m = m0 + 16 * i + ((lane & 16) ? (r + 8) : r);
        int n = n0 + 16 * j + (lane & 15);
        out[(size_t)m * EMBED + n] = acc[i][j][r] + bo[n];
      }
}

// ---------------------------------------------------------------------------
extern "C" void kernel_launch(void* const* d_in, const int* in_sizes, int n_in,
                              void* d_out, int out_size, void* d_ws, size_t ws_size,
                              hipStream_t stream) {
  (void)in_sizes; (void)n_in; (void)out_size; (void)ws_size;

  const float* query = (const float*)d_in[0];
  const float* key   = (const float*)d_in[1];
  const float* value = (const float*)d_in[2];
  const int*   mask  = (const int*)  d_in[3];
  const float* Wq = (const float*)d_in[4];
  const float* bq = (const float*)d_in[5];
  const float* Wk = (const float*)d_in[6];
  const float* bk = (const float*)d_in[7];
  const float* Wv = (const float*)d_in[8];
  const float* bv = (const float*)d_in[9];
  const float* Wo = (const float*)d_in[10];
  const float* bo = (const float*)d_in[11];
  const float* rel = (const float*)d_in[12];

  float* out  = (float*)d_out;
  float* attn = out + (size_t)BATCH * SEQ * EMBED;      // attn_weights region

  // ---- workspace layout (f16 unless noted) ----
  const size_t EE  = (size_t)EMBED * EMBED;             // 1 Mi elems
  const size_t ME  = (size_t)MROWS * EMBED;             // 4 Mi elems
  const size_t BHD = (size_t)BATCH * HEADS * SEQ * HDIM;// 4 Mi elems

  _Float16* WT  = (_Float16*)d_ws;                      // 4 * EE
  _Float16* Xf  = WT  + 4 * EE;                         // 3 * ME
  _Float16* Qh  = Xf  + 3 * ME;                         // BHD  [b][h][s][d]
  _Float16* Kh  = Qh  + BHD;                            // BHD  [b][h][s][d]
  _Float16* VTh = Kh  + BHD;                            // BHD  [b][h][d][s]
  float*    Tb  = (float*)(VTh + BHD);                  // B*H*S*NRELP fp32
  _Float16* ctx = (_Float16*)(Tb + (size_t)BATCH * HEADS * SEQ * NRELP); // ME

  const int CVT_N = (int)ME;
  dim3 cvtGrid((CVT_N + 255) / 256);

  // 1) convert activations to f16
  cvt_f16_kernel<<<cvtGrid, 256, 0, stream>>>(query, Xf + 0 * ME, CVT_N);
  cvt_f16_kernel<<<cvtGrid, 256, 0, stream>>>(key,   Xf + 1 * ME, CVT_N);
  cvt_f16_kernel<<<cvtGrid, 256, 0, stream>>>(value, Xf + 2 * ME, CVT_N);

  // 2) transpose + convert weights
  dim3 twGrid(EMBED / 16, EMBED / 16);
  transpose_w_kernel<<<twGrid, 256, 0, stream>>>(Wq, WT + 0 * EE);
  transpose_w_kernel<<<twGrid, 256, 0, stream>>>(Wk, WT + 1 * EE);
  transpose_w_kernel<<<twGrid, 256, 0, stream>>>(Wv, WT + 2 * EE);
  transpose_w_kernel<<<twGrid, 256, 0, stream>>>(Wo, WT + 3 * EE);

  // 3) projections: (4096/32)*(1024/64) = 2048 waves -> 512 blocks of 4 waves
  proj_gemm_kernel<<<512, 128, 0, stream>>>(Xf + 0 * ME, WT + 0 * EE, bq, Qh, 0);
  proj_gemm_kernel<<<512, 128, 0, stream>>>(Xf + 1 * ME, WT + 1 * EE, bk, Kh, 0);
  proj_gemm_kernel<<<512, 128, 0, stream>>>(Xf + 2 * ME, WT + 2 * EE, bv, VTh, 1);

  // 4) scores = QK^T / 8: per bh (1024/32)*(1024/64)=512 waves -> 128 blocks
  scores_gemm_kernel<<<dim3(128, BATCH * HEADS), 128, 0, stream>>>(Qh, Kh, attn);

  // 5) T = Q @ emb^T: per bh 64 waves of 16 rows -> 16 blocks
  relT_gemm_kernel<<<dim3(16, BATCH * HEADS), 128, 0, stream>>>(Qh, rel, Tb);

  // 6) rel bias + mask + softmax (in place)
  softmax_rel_kernel<<<dim3(SEQ, BATCH * HEADS), 256, 0, stream>>>(attn, Tb, mask);

  // 7) ctx = attn @ V: per bh 1024/32 = 32 waves -> 8 blocks
  pv_gemm_kernel<<<dim3(8, BATCH * HEADS), 128, 0, stream>>>(attn, VTh, ctx);

  // 8) out = ctx @ Wo + bo: 2048 waves -> 512 blocks
  out_gemm_kernel<<<512, 128, 0, stream>>>(ctx, WT + 3 * EE, bo, out);
}
